// Encoder_16922171147127
// MI455X (gfx1250) — compile-verified, matrix operand order checked by reading, not run on hardware
//
#include <hip/hip_runtime.h>
#include <hip/hip_bf16.h>

// ---------------------------------------------------------------------------
// Encoder (L=6, D=512, H=8, DH=64, F=2048, B=2, S=2048) for MI455X / gfx1250.
// Strategy: f16 WMMA (v_wmma_f32_16x16x32_f16) for all GEMMs + flash attention.
// The reference's head split is a plain row-major reshape, so q/k/v buffers
// [B,S,D] are directly [BH=16][S=2048][DH=64] contiguous.
// ---------------------------------------------------------------------------

typedef __attribute__((ext_vector_type(16))) _Float16 v16h;
typedef __attribute__((ext_vector_type(8)))  _Float16 half8;
typedef __attribute__((ext_vector_type(8)))  float    v8f;

#define LCONST 6
#define DCONST 512
#define FCONST 2048
#define SCONST 2048
#define BHCONST 16
#define MCONST 4096            // B*S
#define ATT_SCALE 0.35355339059327379f   // (DH//H)**-0.5 = 8**-0.5

__device__ __forceinline__ v8f vzero8() {
  v8f z;
#pragma unroll
  for (int i = 0; i < 8; ++i) z[i] = 0.0f;
  return z;
}

// Load one f16 WMMA fragment (v16h) as two contiguous 16B chunks.
// ISA f16 A/B layout: element h (0..15) maps to K = (h<8 ? h : h+8) + 8*(lane>>4),
// so per lane the fragment is [base .. base+7] and [base+16 .. base+23] halves,
// where base already includes the 8*(lane>>4) offset.
__device__ __forceinline__ v16h load_frag(const _Float16* base) {
  union { v16h v; half8 h[2]; } u;
  u.h[0] = *(const half8*)(base);
  u.h[1] = *(const half8*)(base + 16);
  return u.v;
}

__device__ __forceinline__ v8f wmma_f16(v16h a, v16h b, v8f c) {
  return __builtin_amdgcn_wmma_f32_16x16x32_f16(
      /*neg_a=*/false, a, /*neg_b=*/false, b,
      /*c_mod=*/(short)0, c, /*reuse_a=*/false, /*reuse_b=*/false);
}

// ---------------------------------------------------------------------------
// Weight convert + transpose: src f32 [K][N] row-major -> dst f16 [N][K]
// ---------------------------------------------------------------------------
__global__ void wtrans_kernel(const float* __restrict__ src,
                              _Float16* __restrict__ dst, int K, int N) {
  int i = blockIdx.x * blockDim.x + threadIdx.x;
  if (i < K * N) {
    int k = i / N, n = i % N;
    dst[(size_t)n * K + k] = (_Float16)src[i];
  }
}

// x init: copy embedding f32 -> running x (d_out, f32) and xh (f16)
__global__ void initx_kernel(const float* __restrict__ src,
                             float* __restrict__ xf,
                             _Float16* __restrict__ xh, int n) {
  int i = blockIdx.x * blockDim.x + threadIdx.x;
  if (i < n) {
    float v = src[i];
    xf[i] = v;
    xh[i] = (_Float16)v;
  }
}

// ---------------------------------------------------------------------------
// GEMM: out[M,N] = act(A[M,K] @ W[K,N] + bias[N])
// A: f16 row-major [M][K]; Wt: f16 pre-transposed [N][K].
// Block: 256 threads = 8 waves; block tile 128x128; wave tile 32x64.
// Fragments loaded straight from global as b128 pairs (L2-resident weights).
// ---------------------------------------------------------------------------
__global__ __launch_bounds__(256) void gemm_wmma_kernel(
    const _Float16* __restrict__ A, const _Float16* __restrict__ Wt,
    const float* __restrict__ bias, _Float16* __restrict__ outH,
    float* __restrict__ outF, int M, int N, int K, int relu) {
  const int lane = threadIdx.x & 31;
  const int w = threadIdx.x >> 5;
  const int wm = w >> 1;          // 0..3
  const int wn = w & 1;           // 0..1
  const int r = lane & 15;
  const int g = lane >> 4;
  const int row0 = blockIdx.y * 128 + wm * 32;
  const int col0 = blockIdx.x * 128 + wn * 64;

  v8f acc[2][4];
#pragma unroll
  for (int mt = 0; mt < 2; ++mt)
#pragma unroll
    for (int nt = 0; nt < 4; ++nt) acc[mt][nt] = vzero8();

  for (int k0 = 0; k0 < K; k0 += 32) {
    v16h a[2], b[4];
#pragma unroll
    for (int mt = 0; mt < 2; ++mt)
      a[mt] = load_frag(A + (size_t)(row0 + mt * 16 + r) * K + k0 + 8 * g);
#pragma unroll
    for (int nt = 0; nt < 4; ++nt)
      b[nt] = load_frag(Wt + (size_t)(col0 + nt * 16 + r) * K + k0 + 8 * g);
    if (k0 + 32 < K)  // hint next A tile into cache (global_prefetch_b8)
      __builtin_prefetch(A + (size_t)(row0 + r) * K + k0 + 32, 0, 1);
#pragma unroll
    for (int mt = 0; mt < 2; ++mt)
#pragma unroll
      for (int nt = 0; nt < 4; ++nt)
        acc[mt][nt] = wmma_f16(a[mt], b[nt], acc[mt][nt]);
  }

  // Epilogue. C layout: VGPR e of lane holds (m = e + 8*g, n = r) per 16x16 tile.
#pragma unroll
  for (int nt = 0; nt < 4; ++nt) {
    const int col = col0 + nt * 16 + r;
    const float bv = bias ? bias[col] : 0.0f;
#pragma unroll
    for (int mt = 0; mt < 2; ++mt) {
#pragma unroll
      for (int e = 0; e < 8; ++e) {
        float val = acc[mt][nt][e] + bv;
        if (relu) val = fmaxf(val, 0.0f);
        const size_t off = (size_t)(row0 + mt * 16 + 8 * g + e) * N + col;
        if (outH) outH[off] = (_Float16)val;
        if (outF) outF[off] = val;
      }
    }
  }
}

// ---------------------------------------------------------------------------
// Flash attention over one head-chunk. q/k/v: f16 [BH][S][DH] (flat), DH=64.
// Grid: (S/16, BH); one wave (32 threads) per block handles 16 query rows,
// streaming keys/values in chunks of 32 with online softmax.
// ---------------------------------------------------------------------------
__global__ __launch_bounds__(32) void flash_attn_kernel(
    const _Float16* __restrict__ q, const _Float16* __restrict__ k,
    const _Float16* __restrict__ v, _Float16* __restrict__ ctx) {
  const int bh = blockIdx.y;
  const int qt = blockIdx.x;            // 16-row query tile
  const int lane = threadIdx.x;
  const int r = lane & 15;
  const int g = lane >> 4;

  __shared__ alignas(16) _Float16 Vt[64][32];  // [dh][key] transposed V tile
  __shared__ alignas(16) _Float16 P[16][32];   // probs, D-layout -> A-layout bridge

  const _Float16* Qb = q + ((size_t)bh * SCONST + qt * 16) * 64;
  v16h aq[2];
#pragma unroll
  for (int s = 0; s < 2; ++s)
    aq[s] = load_frag(Qb + (size_t)r * 64 + s * 32 + 8 * g);

  v8f O[4];
#pragma unroll
  for (int i = 0; i < 4; ++i) O[i] = vzero8();
  float m_[8], l_[8];
#pragma unroll
  for (int e = 0; e < 8; ++e) { m_[e] = -1.0e30f; l_[e] = 0.0f; }

  for (int kt = 0; kt < SCONST / 32; ++kt) {
    // Stage V tile transposed: Vt[dh][key]
    const _Float16* Vb = v + ((size_t)bh * SCONST + kt * 32) * 64;
#pragma unroll
    for (int c = 0; c < 8; ++c) {
      const int idx = c * 32 + lane;       // 0..255 b128 chunks
      const int key = idx >> 3;            // 0..31
      const int dh0 = (idx & 7) * 8;
      half8 t = *(const half8*)(Vb + (size_t)key * 64 + dh0);
#pragma unroll
      for (int j = 0; j < 8; ++j) Vt[dh0 + j][key] = t[j];
    }

    // S = Q @ K^T (16 x 32) over DH=64 (two k-slices), two 16-key n-tiles
    const _Float16* Kb = k + ((size_t)bh * SCONST + kt * 32) * 64;
    v8f c0 = vzero8(), c1 = vzero8();
#pragma unroll
    for (int s = 0; s < 2; ++s) {
      v16h b0 = load_frag(Kb + (size_t)(r) * 64 + s * 32 + 8 * g);
      v16h b1 = load_frag(Kb + (size_t)(16 + r) * 64 + s * 32 + 8 * g);
      c0 = wmma_f16(aq[s], b0, c0);
      c1 = wmma_f16(aq[s], b1, c1);
    }

    // Online softmax. Row m = e + 8*g lives on lanes with the same g;
    // shfl_xor masks 1/2/4/8 reduce across the 16 n-lanes of that row.
#pragma unroll
    for (int e = 0; e < 8; ++e) {
      float s0 = c0[e] * ATT_SCALE;
      float s1 = c1[e] * ATT_SCALE;
      float tm = fmaxf(s0, s1);
#pragma unroll
      for (int msk = 1; msk < 16; msk <<= 1) tm = fmaxf(tm, __shfl_xor(tm, msk, 32));
      const float mn = fmaxf(m_[e], tm);
      const float corr = __expf(m_[e] - mn);
      m_[e] = mn;
      const float p0 = __expf(s0 - mn);
      const float p1 = __expf(s1 - mn);
      c0[e] = p0;
      c1[e] = p1;
      float rs = p0 + p1;
#pragma unroll
      for (int msk = 1; msk < 16; msk <<= 1) rs += __shfl_xor(rs, msk, 32);
      l_[e] = l_[e] * corr + rs;
      O[0][e] *= corr; O[1][e] *= corr; O[2][e] *= corr; O[3][e] *= corr;
    }

    // D-layout -> LDS -> A-layout for the PV matmul
#pragma unroll
    for (int e = 0; e < 8; ++e) {
      P[8 * g + e][r]      = (_Float16)c0[e];
      P[8 * g + e][16 + r] = (_Float16)c1[e];
    }
    v16h ap = load_frag(&P[r][8 * g]);
#pragma unroll
    for (int nt = 0; nt < 4; ++nt) {
      v16h bv = load_frag(&Vt[nt * 16 + r][8 * g]);
      O[nt] = wmma_f16(ap, bv, O[nt]);
    }
  }

  // Normalize and store ctx as f16 (flat [BH][S][DH] == [B,S,D])
#pragma unroll
  for (int nt = 0; nt < 4; ++nt) {
#pragma unroll
    for (int e = 0; e < 8; ++e) {
      const float val = O[nt][e] / l_[e];
      const int row = qt * 16 + 8 * g + e;
      ctx[((size_t)bh * SCONST + row) * 64 + nt * 16 + r] = (_Float16)val;
    }
  }
}

// ---------------------------------------------------------------------------
// Fused residual + LayerNorm: y = LN(xres + proj) * gamma + beta
// Writes f32 (running x, in d_out) and f16 (for next GEMM). One block per row.
// ---------------------------------------------------------------------------
__global__ __launch_bounds__(128) void add_ln_kernel(
    const float* __restrict__ xres, const float* __restrict__ proj,
    const float* __restrict__ gamma, const float* __restrict__ beta,
    float* __restrict__ xout, _Float16* __restrict__ xh) {
  const int row = blockIdx.x;
  const int tid = threadIdx.x;
  const int lane = tid & 31;
  const int wv = tid >> 5;
  const size_t base = (size_t)row * DCONST;

  float vals[4];
  float s = 0.0f, ss = 0.0f;
#pragma unroll
  for (int j = 0; j < 4; ++j) {
    const int c = tid + j * 128;
    const float vv = xres[base + c] + proj[base + c];
    vals[j] = vv;
    s += vv;
    ss += vv * vv;
  }
#pragma unroll
  for (int msk = 1; msk < 32; msk <<= 1) {
    s += __shfl_xor(s, msk, 32);
    ss += __shfl_xor(ss, msk, 32);
  }
  __shared__ float sb[4], sb2[4];
  if (lane == 0) { sb[wv] = s; sb2[wv] = ss; }
  __syncthreads();
  s = sb[0] + sb[1] + sb[2] + sb[3];
  ss = sb2[0] + sb2[1] + sb2[2] + sb2[3];
  const float mean = s * (1.0f / DCONST);
  const float var = ss * (1.0f / DCONST) - mean * mean;
  const float rstd = rsqrtf(var + 1e-5f);
#pragma unroll
  for (int j = 0; j < 4; ++j) {
    const int c = tid + j * 128;
    const float y = (vals[j] - mean) * rstd * gamma[c] + beta[c];
    xout[base + c] = y;
    xh[base + c] = (_Float16)y;
  }
}

// ---------------------------------------------------------------------------
// Host launcher
// ---------------------------------------------------------------------------
extern "C" void kernel_launch(void* const* d_in, const int* in_sizes, int n_in,
                              void* d_out, int out_size, void* d_ws, size_t ws_size,
                              hipStream_t stream) {
  (void)in_sizes; (void)n_in; (void)out_size; (void)ws_size;
  const int D = DCONST, F = FCONST, M = MCONST;

  const float* x_in = (const float*)d_in[0];
  // d_in[1] = inputs_len : unused by the reference math
  const float* Wq = (const float*)d_in[2];
  const float* bq = (const float*)d_in[3];
  const float* Wk = (const float*)d_in[4];
  const float* bk = (const float*)d_in[5];
  const float* Wv = (const float*)d_in[6];
  const float* bv = (const float*)d_in[7];
  const float* Wo = (const float*)d_in[8];
  const float* bo = (const float*)d_in[9];
  const float* g1 = (const float*)d_in[10];
  const float* b1 = (const float*)d_in[11];
  const float* W1 = (const float*)d_in[12];
  const float* c1 = (const float*)d_in[13];
  const float* W2 = (const float*)d_in[14];
  const float* c2 = (const float*)d_in[15];
  const float* g2 = (const float*)d_in[16];
  const float* b2 = (const float*)d_in[17];

  float* xf = (float*)d_out;  // running x (f32) lives in d_out

  // Workspace carve-up (halves). Total ~82 MB.
  _Float16* ws = (_Float16*)d_ws;
  size_t o = 0;
  _Float16* wqT = ws + o; o += (size_t)LCONST * D * D;
  _Float16* wkT = ws + o; o += (size_t)LCONST * D * D;
  _Float16* wvT = ws + o; o += (size_t)LCONST * D * D;
  _Float16* woT = ws + o; o += (size_t)LCONST * D * D;
  _Float16* w1T = ws + o; o += (size_t)LCONST * D * F;   // [F][D] per layer
  _Float16* w2T = ws + o; o += (size_t)LCONST * F * D;   // [D][F] per layer
  _Float16* xh  = ws + o; o += (size_t)M * D;
  _Float16* qh  = ws + o; o += (size_t)M * D;
  _Float16* kh  = ws + o; o += (size_t)M * D;
  _Float16* vh  = ws + o; o += (size_t)M * D;
  _Float16* cth = ws + o; o += (size_t)M * D;
  _Float16* hh  = ws + o; o += (size_t)M * F;
  float* tmpf = (float*)(ws + o);                        // f32 scratch [M][D]

  const int thr = 256;
  // Weight convert + transpose (per launch; deterministic, no caching)
  for (int l = 0; l < LCONST; ++l) {
    const int nDD = D * D, nDF = D * F;
    wtrans_kernel<<<(nDD + thr - 1) / thr, thr, 0, stream>>>(Wq + (size_t)l * nDD, wqT + (size_t)l * nDD, D, D);
    wtrans_kernel<<<(nDD + thr - 1) / thr, thr, 0, stream>>>(Wk + (size_t)l * nDD, wkT + (size_t)l * nDD, D, D);
    wtrans_kernel<<<(nDD + thr - 1) / thr, thr, 0, stream>>>(Wv + (size_t)l * nDD, wvT + (size_t)l * nDD, D, D);
    wtrans_kernel<<<(nDD + thr - 1) / thr, thr, 0, stream>>>(Wo + (size_t)l * nDD, woT + (size_t)l * nDD, D, D);
    wtrans_kernel<<<(nDF + thr - 1) / thr, thr, 0, stream>>>(W1 + (size_t)l * nDF, w1T + (size_t)l * nDF, D, F);
    wtrans_kernel<<<(nDF + thr - 1) / thr, thr, 0, stream>>>(W2 + (size_t)l * nDF, w2T + (size_t)l * nDF, F, D);
  }

  const int nx = M * D;
  initx_kernel<<<(nx + thr - 1) / thr, thr, 0, stream>>>(x_in, xf, xh, nx);

  const dim3 gP(D / 128, M / 128);   // N=512 GEMMs
  const dim3 gF1(F / 128, M / 128);  // N=2048 GEMM
  const dim3 gAtt(SCONST / 16, BHCONST);

  for (int l = 0; l < LCONST; ++l) {
    const size_t oDD = (size_t)l * D * D, oDF = (size_t)l * D * F;
    // Q, K, V projections (f16 out)
    gemm_wmma_kernel<<<gP, 256, 0, stream>>>(xh, wqT + oDD, bq + (size_t)l * D, qh, nullptr, M, D, D, 0);
    gemm_wmma_kernel<<<gP, 256, 0, stream>>>(xh, wkT + oDD, bk + (size_t)l * D, kh, nullptr, M, D, D, 0);
    gemm_wmma_kernel<<<gP, 256, 0, stream>>>(xh, wvT + oDD, bv + (size_t)l * D, vh, nullptr, M, D, D, 0);
    // Attention (16 heads of [2048,64] thanks to the row-major reshape quirk)
    flash_attn_kernel<<<gAtt, 32, 0, stream>>>(qh, kh, vh, cth);
    // Output projection (f32 out) + residual LN -> x, xh
    gemm_wmma_kernel<<<gP, 256, 0, stream>>>(cth, woT + oDD, bo + (size_t)l * D, nullptr, tmpf, M, D, D, 0);
    add_ln_kernel<<<M, 128, 0, stream>>>(xf, tmpf, g1 + (size_t)l * D, b1 + (size_t)l * D, xf, xh);
    // FFN: relu(x@W1+c1) @ W2 + c2, then residual LN
    gemm_wmma_kernel<<<gF1, 256, 0, stream>>>(xh, w1T + oDF, c1 + (size_t)l * F, hh, nullptr, M, F, D, 1);
    gemm_wmma_kernel<<<gP, 256, 0, stream>>>(hh, w2T + oDF, c2 + (size_t)l * D, nullptr, tmpf, M, D, F, 0);
    add_ln_kernel<<<M, 128, 0, stream>>>(xf, tmpf, g2 + (size_t)l * D, b2 + (size_t)l * D, xf, xh);
  }
}
